// CausalSelfAttention_7499012899210
// MI455X (gfx1250) — matrix operand order repaired and checked by reading, not compiled, not run on hardware
//
#include <hip/hip_runtime.h>
#include <hip/hip_bf16.h>

// CDNA5 / gfx1250 wave32 WMMA implementation of ternary-quantized causal self-attention.
// Pipeline: quantize(w*)->bf16, x->bf16, 3 WMMA GEMMs (q,k,v) with TDM tile staging,
// RoPE(q,k), flash-attention (bf16 WMMA, online softmax), output WMMA GEMM -> f32.

typedef __bf16 bf16_t;
typedef __attribute__((ext_vector_type(16))) __bf16 v16bf;
typedef __attribute__((ext_vector_type(8)))  float  v8f;
typedef unsigned int u32x4 __attribute__((ext_vector_type(4)));
typedef int          i32x4 __attribute__((ext_vector_type(4)));
typedef int          i32x8 __attribute__((ext_vector_type(8)));

#define B_   2
#define T_   2048
#define C_   1024
#define H_   16
#define HD_  64
#define M_   (B_ * T_)        // 4096 rows of x
#define GRP_ 128              // ternary quant group size

#if __has_builtin(__builtin_amdgcn_tensor_load_to_lds)
#define HAVE_TDM 1
#else
#define HAVE_TDM 0
#endif

union AFrag { v16bf v; float4 q[2]; };

// A-matrix fragment, 16x32 bf16 (M x K), row-major source with leading dim lda (elems).
// Contiguous k order used here (same as B frags): lanes 0-15 hold rows M=lane, K=0..15; lanes 16-31 K=16..31.
__device__ __forceinline__ v16bf load_frag_a(const bf16_t* base, int lda, int row0, int k0) {
  int lane = threadIdx.x & 31;
  int m    = lane & 15;
  int kb   = (lane >> 4) * 16;
  const bf16_t* p = base + (size_t)(row0 + m) * lda + (k0 + kb);
  AFrag f;
  f.q[0] = *(const float4*)(p);        // K = kb .. kb+7
  f.q[1] = *(const float4*)(p + 8);    // K = kb+8 .. kb+15
  return f.v;
}

// B-matrix fragment, 32x16 bf16 (K x N) where B[k][n] = W[col0+n][k0+k] (row-major W).
// Contiguous k order used here (same as A frags): lane n (0-15) holds column n, K=0..15; lanes 16-31 hold K=16..31.
__device__ __forceinline__ v16bf load_frag_b(const bf16_t* base, int ldb, int col0, int k0) {
  int lane = threadIdx.x & 31;
  int n    = lane & 15;
  int kb   = (lane >> 4) * 16;
  return *(const v16bf*)(base + (size_t)(col0 + n) * ldb + (k0 + kb));
}

__device__ __forceinline__ v8f wmma_bf16(v16bf a, v16bf b, v8f c) {
  return __builtin_amdgcn_wmma_f32_16x16x32_bf16(false, a, false, b, (short)0, c, false, false);
}

#if HAVE_TDM
// Issue one TDM 2D tile load: tile_y rows of tile_x bf16 elements, global row stride
// row_stride_elems, into LDS at byte offset lds_off with hardware padding
// (pad_amount DWORDs inserted after every 2^(pad_interval+1) DWORDs stored).
__device__ __forceinline__ void tdm_load_tile_2d(unsigned lds_off, const void* gptr,
                                                 int tile_x, int tile_y,
                                                 unsigned row_stride_elems,
                                                 int pad_interval_code, int pad_amount_code) {
  unsigned long long ga = (unsigned long long)(uintptr_t)gptr;
  u32x4 g0 = {};
  g0[0] = 1u;                                               // count=1, user descriptor
  g0[1] = lds_off;                                          // lds_addr (bytes)
  g0[2] = (unsigned)ga;                                     // global_addr[31:0]
  g0[3] = (unsigned)((ga >> 32) & 0x01ffffffu) | (2u << 30);// global_addr[56:32], type=2
  i32x8 g1 = {};
  // data_size=1 (2 bytes), pad_enable, pad_interval, pad_amount; workgroup_mask=0
  g1[0] = (1 << 16) | (1 << 20) | (pad_interval_code << 22) | (pad_amount_code << 25);
  // tensor_dim0 = tensor_dim1 = 1<<20 (large; no OOB clipping for our in-range tiles)
  g1[1] = 0;                                                // abar=0, dim0[15:0]=0
  g1[2] = 16;                                               // dim0[31:16]=16, dim1[15:0]=0
  g1[3] = 16 | (tile_x << 16);                              // dim1[31:16]=16, tile_dim0
  g1[4] = tile_y & 0xffff;                                  // tile_dim1; tile_dim2=0
  g1[5] = (int)row_stride_elems;                            // tensor_dim0_stride[31:0]
  g1[6] = 0;                                                // stride hi bits, dim1_stride lo
  g1[7] = 0;
  i32x4 gz4 = {};
  i32x8 gz8 = {};
  // 6-arg form (clang-23 / therock-10.0 headers): (g0, g1, g2, g3, extra, cpol)
  __builtin_amdgcn_tensor_load_to_lds(g0, g1, gz4, gz4, gz8, 0);
}
#endif

// ---------------- elementwise f32 -> bf16 ----------------
__global__ __launch_bounds__(256) void f32_to_bf16_kernel(const float* __restrict__ in,
                                                          bf16_t* __restrict__ out, int n) {
  int i = blockIdx.x * blockDim.x + threadIdx.x;
  if (i < n) out[i] = (bf16_t)in[i];
}

// ---------------- ternary quantize: one wave per 128-elem group ----------------
__global__ __launch_bounds__(256) void quantize_ternary_kernel(const float* __restrict__ W,
                                                               bf16_t* __restrict__ Wq,
                                                               int n_groups) {
  int wave = (int)((blockIdx.x * blockDim.x + threadIdx.x) >> 5);
  int lane = threadIdx.x & 31;
  if (wave >= n_groups) return;
  const float* g = W + (size_t)wave * GRP_;
  float v[4];
  float s = 0.f;
  #pragma unroll
  for (int i = 0; i < 4; ++i) { v[i] = g[lane + 32 * i]; s += fabsf(v[i]); }
  #pragma unroll
  for (int off = 16; off >= 1; off >>= 1) s += __shfl_xor(s, off, 32);
  float scale = fmaxf(s * (1.0f / (float)GRP_), 1e-8f);
  float inv = 1.0f / scale;
  bf16_t* o = Wq + (size_t)wave * GRP_;
  #pragma unroll
  for (int i = 0; i < 4; ++i) {
    float wn = v[i] * inv;
    float q = (wn > 0.5f) ? 1.0f : ((wn < -0.5f) ? -1.0f : 0.0f);
    o[lane + 32 * i] = (bf16_t)(q * scale);
  }
}

// ---------------- GEMM: out[m,n] = sum_k A[m,k] * W[n,k]  (X @ W^T) ----------------
// Block: 256 threads = 8 waves arranged 2(M)x4(N); wave tile 64x32; block tile 128x128.
// TDM path: wave 0 DMAs the 128x64 A/B tiles into padded LDS each K-step (TENSORcnt),
// all waves compute WMMA fragments from LDS.
template <int F32OUT>
__global__ __launch_bounds__(256) void gemm_xwT(const bf16_t* __restrict__ A,
                                                const bf16_t* __restrict__ W,
                                                void* __restrict__ outp,
                                                int Ndim, int Kdim) {
  int wave  = threadIdx.x >> 5;
  int lane  = threadIdx.x & 31;
  int row0w = (wave >> 2) * 64;
  int col0w = (wave & 3) * 32;
  int rowB  = blockIdx.y * 128;
  int colB  = blockIdx.x * 128;

  v8f acc[4][2] = {};

#if HAVE_TDM
  // 64 K elems + 16-elem (32B) hw pad per row -> 160B row stride (32B aligned, 2-way banks)
  __shared__ __align__(32) bf16_t lta[128][80];
  __shared__ __align__(32) bf16_t ltb[128][80];
  unsigned offa = (unsigned)(uintptr_t)&lta[0][0];
  unsigned offb = (unsigned)(uintptr_t)&ltb[0][0];
  for (int k0 = 0; k0 < Kdim; k0 += 64) {
    if (wave == 0) {
      tdm_load_tile_2d(offa, A + (size_t)rowB * Kdim + k0, 64, 128, (unsigned)Kdim, 4, 7);
      tdm_load_tile_2d(offb, W + (size_t)colB * Kdim + k0, 64, 128, (unsigned)Kdim, 4, 7);
      __builtin_amdgcn_s_wait_tensorcnt((short)0);
    }
    __syncthreads();
    #pragma unroll
    for (int kf = 0; kf < 2; ++kf) {
      v16bf a[4], b[2];
      #pragma unroll
      for (int i = 0; i < 4; ++i) a[i] = load_frag_a(&lta[0][0], 80, row0w + 16 * i, kf * 32);
      #pragma unroll
      for (int j = 0; j < 2; ++j) b[j] = load_frag_b(&ltb[0][0], 80, col0w + 16 * j, kf * 32);
      #pragma unroll
      for (int i = 0; i < 4; ++i)
        #pragma unroll
        for (int j = 0; j < 2; ++j)
          acc[i][j] = wmma_bf16(a[i], b[j], acc[i][j]);
    }
    __syncthreads();
  }
#else
  for (int k0 = 0; k0 < Kdim; k0 += 32) {
    v16bf a[4], b[2];
    #pragma unroll
    for (int i = 0; i < 4; ++i) a[i] = load_frag_a(A, Kdim, rowB + row0w + 16 * i, k0);
    #pragma unroll
    for (int j = 0; j < 2; ++j) b[j] = load_frag_b(W, Kdim, colB + col0w + 16 * j, k0);
    #pragma unroll
    for (int i = 0; i < 4; ++i)
      #pragma unroll
      for (int j = 0; j < 2; ++j)
        acc[i][j] = wmma_bf16(a[i], b[j], acc[i][j]);
  }
#endif

  // C/D layout: lane n = lane&15, rows m = (lane>>4)*8 + r
  int n  = lane & 15;
  int mo = (lane >> 4) * 8;
  #pragma unroll
  for (int i = 0; i < 4; ++i)
    #pragma unroll
    for (int j = 0; j < 2; ++j)
      #pragma unroll
      for (int r = 0; r < 8; ++r) {
        size_t off = (size_t)(rowB + row0w + 16 * i + mo + r) * Ndim + (colB + col0w + 16 * j + n);
        if (F32OUT) ((float*)outp)[off] = acc[i][j][r];
        else        ((bf16_t*)outp)[off] = (bf16_t)acc[i][j][r];
      }
}

// ---------------- RoPE in place on bf16 [B,T,C]; pairs (d, d+32) per head ----------------
__global__ __launch_bounds__(256) void rope_kernel(bf16_t* __restrict__ q,
                                                   bf16_t* __restrict__ k,
                                                   const float* __restrict__ cs,
                                                   const float* __restrict__ sn) {
  int idx = blockIdx.x * blockDim.x + threadIdx.x;   // B*T*H*32 = 2M threads
  int d = idx & 31;
  int h = (idx >> 5) & 15;
  int t = (idx >> 9) & (T_ - 1);
  int b = idx >> 20;
  if (b >= B_) return;
  size_t base = ((size_t)(b * T_ + t)) * C_ + h * HD_;
  float c = cs[t * 32 + d];
  float s = sn[t * 32 + d];
  float x1 = (float)q[base + d], x2 = (float)q[base + 32 + d];
  q[base + d]      = (bf16_t)(x1 * c - x2 * s);
  q[base + 32 + d] = (bf16_t)(x1 * s + x2 * c);
  x1 = (float)k[base + d]; x2 = (float)k[base + 32 + d];
  k[base + d]      = (bf16_t)(x1 * c - x2 * s);
  k[base + 32 + d] = (bf16_t)(x1 * s + x2 * c);
}

// ---------------- flash attention: block = (qtile 64 rows, one (b,h)); 4 waves x 16 rows ----
__global__ __launch_bounds__(128) void flash_attn_kernel(const bf16_t* __restrict__ qg,
                                                         const bf16_t* __restrict__ kg,
                                                         const bf16_t* __restrict__ vg,
                                                         bf16_t* __restrict__ yg) {
  __shared__ __align__(32) bf16_t vt[HD_][80];       // V^T tile: vt[d][key], padded rows
  __shared__ __align__(32) bf16_t pl[4][16][64];     // per-wave P tile staging

  const float NEG_INF = -__builtin_inff();
  int qtile = blockIdx.x;          // 0..31
  int bh    = blockIdx.y;          // 0..31
  int b = bh >> 4, h = bh & 15;
  const bf16_t* qb = qg + (size_t)b * T_ * C_;
  const bf16_t* kb = kg + (size_t)b * T_ * C_;
  const bf16_t* vb = vg + (size_t)b * T_ * C_;
  bf16_t*       yb = yg + (size_t)b * T_ * C_;

  int wave  = threadIdx.x >> 5;
  int lane  = threadIdx.x & 31;
  int nlane = lane & 15;
  int mo    = (lane >> 4) * 8;
  int row0  = qtile * 64 + wave * 16;
  int colh  = h * HD_;

  // Q fragments resident for all key tiles (16 rows x 64 d = 2 A-frags)
  v16bf qf[2];
  qf[0] = load_frag_a(qb, C_, row0, colh);
  qf[1] = load_frag_a(qb, C_, row0, colh + 32);

  v8f o[4] = {};                    // 16x64 output accum (4 d-frags)
  float mstate[8], lstate[8];
  #pragma unroll
  for (int r = 0; r < 8; ++r) { mstate[r] = NEG_INF; lstate[r] = 0.f; }

  for (int kt = 0; kt <= qtile; ++kt) {
    __syncthreads();                               // prior PV done before vt overwrite
    for (int idx = threadIdx.x; idx < 64 * 64; idx += 128) {
      int kk = idx >> 6, d = idx & 63;
      vt[d][kk] = vb[(size_t)(kt * 64 + kk) * C_ + colh + d];
    }
    __syncthreads();

    // S = Q K^T  (keys as B-matrix: key rows contiguous in d -> direct global frags)
    v8f s4[4] = {};
    #pragma unroll
    for (int kf = 0; kf < 2; ++kf) {
      v16bf bfr[4];
      #pragma unroll
      for (int nf = 0; nf < 4; ++nf)
        bfr[nf] = load_frag_b(kb, C_, kt * 64 + nf * 16, colh + kf * 32);
      #pragma unroll
      for (int nf = 0; nf < 4; ++nf) s4[nf] = wmma_bf16(qf[kf], bfr[nf], s4[nf]);
    }

    // scale + causal mask (diagonal tile only)
    #pragma unroll
    for (int nf = 0; nf < 4; ++nf)
      #pragma unroll
      for (int r = 0; r < 8; ++r) {
        float sv = s4[nf][r] * 0.125f;             // Hd^-0.5 = 1/8
        if (kt == qtile) {
          int ng = kt * 64 + nf * 16 + nlane;
          int mg = row0 + mo + r;
          if (ng > mg) sv = NEG_INF;
        }
        s4[nf][r] = sv;
      }

    // online softmax: rows live on 16 lanes (xor masks 1,2,4,8 stay in the N-group)
    #pragma unroll
    for (int r = 0; r < 8; ++r) {
      float rmax = s4[0][r];
      #pragma unroll
      for (int nf = 1; nf < 4; ++nf) rmax = fmaxf(rmax, s4[nf][r]);
      #pragma unroll
      for (int msk = 8; msk >= 1; msk >>= 1) rmax = fmaxf(rmax, __shfl_xor(rmax, msk, 32));
      float mnew = fmaxf(mstate[r], rmax);
      float corr = __expf(mstate[r] - mnew);
      float rsum = 0.f;
      #pragma unroll
      for (int nf = 0; nf < 4; ++nf) {
        float p = __expf(s4[nf][r] - mnew);
        s4[nf][r] = p;
        rsum += p;
      }
      #pragma unroll
      for (int msk = 8; msk >= 1; msk >>= 1) rsum += __shfl_xor(rsum, msk, 32);
      lstate[r] = lstate[r] * corr + rsum;
      mstate[r] = mnew;
      #pragma unroll
      for (int df = 0; df < 4; ++df) o[df][r] *= corr;
    }

    // P: C-layout -> LDS -> A-layout
    #pragma unroll
    for (int nf = 0; nf < 4; ++nf)
      #pragma unroll
      for (int r = 0; r < 8; ++r)
        pl[wave][mo + r][nf * 16 + nlane] = (bf16_t)s4[nf][r];
    __syncthreads();

    // O += P V  (B-frags contiguous over keys in the transposed vt tile)
    #pragma unroll
    for (int kf = 0; kf < 2; ++kf) {
      v16bf pa = load_frag_a(&pl[wave][0][0], 64, 0, kf * 32);
      #pragma unroll
      for (int df = 0; df < 4; ++df) {
        v16bf vf = load_frag_b(&vt[0][0], 80, df * 16, kf * 32);
        o[df] = wmma_bf16(pa, vf, o[df]);
      }
    }
  }

  #pragma unroll
  for (int df = 0; df < 4; ++df)
    #pragma unroll
    for (int r = 0; r < 8; ++r) {
      float ov = o[df][r] / lstate[r];
      yb[(size_t)(row0 + mo + r) * C_ + colh + df * 16 + nlane] = (bf16_t)ov;
    }
}

// ---------------- host launch ----------------
extern "C" void kernel_launch(void* const* d_in, const int* in_sizes, int n_in,
                              void* d_out, int out_size, void* d_ws, size_t ws_size,
                              hipStream_t stream) {
  (void)in_sizes; (void)n_in; (void)out_size; (void)ws_size;
  const float* x  = (const float*)d_in[0];
  const float* wq = (const float*)d_in[1];
  const float* wk = (const float*)d_in[2];
  const float* wv = (const float*)d_in[3];
  const float* wo = (const float*)d_in[4];
  const float* rc = (const float*)d_in[5];
  const float* rs = (const float*)d_in[6];

  // workspace carve-up (bf16): x, 4 quantized weights, q, k, v, y  (~48 MB total)
  bf16_t* xb  = (bf16_t*)d_ws;
  bf16_t* wqb = xb  + (size_t)M_ * C_;
  bf16_t* wkb = wqb + (size_t)C_ * C_;
  bf16_t* wvb = wkb + (size_t)C_ * C_;
  bf16_t* wob = wvb + (size_t)C_ * C_;
  bf16_t* qb  = wob + (size_t)C_ * C_;
  bf16_t* kbf = qb  + (size_t)M_ * C_;
  bf16_t* vbf = kbf + (size_t)M_ * C_;
  bf16_t* ybf = vbf + (size_t)M_ * C_;

  int nx = M_ * C_;                               // 4,194,304
  f32_to_bf16_kernel<<<(nx + 255) / 256, 256, 0, stream>>>(x, xb, nx);

  int n_groups = C_ * C_ / GRP_;                  // 8192 groups per matrix
  int qgrid = (n_groups * 32 + 255) / 256;
  quantize_ternary_kernel<<<qgrid, 256, 0, stream>>>(wq, wqb, n_groups);
  quantize_ternary_kernel<<<qgrid, 256, 0, stream>>>(wk, wkb, n_groups);
  quantize_ternary_kernel<<<qgrid, 256, 0, stream>>>(wv, wvb, n_groups);
  quantize_ternary_kernel<<<qgrid, 256, 0, stream>>>(wo, wob, n_groups);

  dim3 ggrid(C_ / 128, M_ / 128);                 // (8, 32)
  gemm_xwT<0><<<ggrid, 256, 0, stream>>>(xb, wqb, (void*)qb,  C_, C_);
  gemm_xwT<0><<<ggrid, 256, 0, stream>>>(xb, wkb, (void*)kbf, C_, C_);
  gemm_xwT<0><<<ggrid, 256, 0, stream>>>(xb, wvb, (void*)vbf, C_, C_);

  int nr = B_ * T_ * H_ * 32;                     // 2,097,152
  rope_kernel<<<(nr + 255) / 256, 256, 0, stream>>>(qb, kbf, rc, rs);

  dim3 agrid(T_ / 64, B_ * H_);                   // (32, 32)
  flash_attn_kernel<<<agrid, 128, 0, stream>>>(qb, kbf, vbf, ybf);

  gemm_xwT<1><<<ggrid, 256, 0, stream>>>(ybf, wob, d_out, C_, C_);
}
